// VQTransformer_33724083208260
// MI455X (gfx1250) — compile-verified, matrix operand order checked by reading, not run on hardware
//
#include <hip/hip_runtime.h>
#include <hip/hip_bf16.h>

// ---------------- model constants ----------------
#define B_   4
#define C_   256
#define NH_  8
#define HD_  32
#define DC_  1024
#define NT_  256                 // tokens per image (16x16)
#define L_   4
#define V_   1024
#define MR_  (B_*NT_)            // 1024 rows (all batches concatenated)
#define SCALE_ 0.17677669529663687f   // 1/sqrt(32)

typedef __attribute__((ext_vector_type(16))) _Float16 v16h;
typedef __attribute__((ext_vector_type(8)))  float    v8f;

// ---------------- WMMA core helpers ----------------
__device__ __forceinline__ v8f wmma_f16(v16h a, v16h b, v8f c) {
  return __builtin_amdgcn_wmma_f32_16x16x32_f16(false, a, false, b, (short)0, c, false, false);
}

// A fragment, 16x32 f16, row-major A[m][k], row stride lda (halfs). Contiguous k -> 2x b128.
__device__ __forceinline__ v16h frag_a_f16(const _Float16* __restrict__ A, int lda, int lane) {
  const int r = lane & 15, g = (lane >> 4) & 1;
  const _Float16* p = A + (long)r * lda + g * 8;
  v16h a;
#pragma unroll
  for (int i = 0; i < 8; ++i) a[i] = p[i];
#pragma unroll
  for (int i = 0; i < 8; ++i) a[8 + i] = p[16 + i];
  return a;
}

// B fragment, 32x16, element B[k][n] at Bp + k + n*ldn  (k contiguous -> 2x b128 per lane)
__device__ __forceinline__ v16h frag_b_k1(const _Float16* __restrict__ Bp, long ldn, int lane) {
  const int n = lane & 15, g = (lane >> 4) & 1;
  const _Float16* p = Bp + (long)n * ldn + g * 8;
  v16h b;
#pragma unroll
  for (int i = 0; i < 8; ++i) b[i] = p[i];
#pragma unroll
  for (int i = 0; i < 8; ++i) b[8 + i] = p[16 + i];
  return b;
}

__device__ __forceinline__ float gelu_exact(float x) {
  return 0.5f * x * (1.0f + erff(x * 0.70710678118654752f));
}

// ------- generic linear GEMM, 2x2 blocked (32x32 per wave), all-contiguous fragments -------
// Y = A(f16,MxK) * B(f16, stored [n][k], ldn = row pitch) + bias
// flags: 1 = write f32, 2 = accumulate into f32, 4 = write f16 row-major,
//        8 = write f16 transposed (oh[col*Mrows + row])
__global__ void k_gemm_lin(const _Float16* __restrict__ A, int lda,
                           const _Float16* __restrict__ W, long ldn,
                           const float* __restrict__ bias,
                           float* __restrict__ of, _Float16* __restrict__ oh,
                           int K, int Ncols, int Mrows, int flags) {
  const int lane = threadIdx.x;
  const int tn = blockIdx.x, tm = blockIdx.y;
  const _Float16* Ap0 = A + (long)tm * 32 * lda;
  const _Float16* Ap1 = Ap0 + (long)16 * lda;
  const _Float16* Wp0 = W + (long)tn * 32 * ldn;
  const _Float16* Wp1 = Wp0 + (long)16 * ldn;
  v8f acc00 = {0.f,0.f,0.f,0.f,0.f,0.f,0.f,0.f};
  v8f acc01 = acc00, acc10 = acc00, acc11 = acc00;
  for (int k = 0; k < K; k += 32) {
    const v16h a0 = frag_a_f16(Ap0 + k, lda, lane);
    const v16h a1 = frag_a_f16(Ap1 + k, lda, lane);
    const v16h b0 = frag_b_k1(Wp0 + k, ldn, lane);
    const v16h b1 = frag_b_k1(Wp1 + k, ldn, lane);
    acc00 = wmma_f16(a0, b0, acc00);
    acc01 = wmma_f16(a0, b1, acc01);
    acc10 = wmma_f16(a1, b0, acc10);
    acc11 = wmma_f16(a1, b1, acc11);
  }
  const int nl = lane & 15, g = (lane >> 4) & 1;
  const int col0 = tn * 32 + nl, col1 = col0 + 16;
  const float bv0 = bias ? bias[col0] : 0.0f;
  const float bv1 = bias ? bias[col1] : 0.0f;
#pragma unroll
  for (int i = 0; i < 8; ++i) {
    const long row0 = (long)tm * 32 + g * 8 + i;
    const long row1 = row0 + 16;
    float v00 = acc00[i] + bv0, v01 = acc01[i] + bv1;
    float v10 = acc10[i] + bv0, v11 = acc11[i] + bv1;
    const long i00 = row0 * Ncols + col0, i01 = row0 * Ncols + col1;
    const long i10 = row1 * Ncols + col0, i11 = row1 * Ncols + col1;
    if (flags & 2) { v00 += of[i00]; v01 += of[i01]; v10 += of[i10]; v11 += of[i11]; }
    if (flags & 1) { of[i00] = v00; of[i01] = v01; of[i10] = v10; of[i11] = v11; }
    if (flags & 4) { oh[i00] = (_Float16)v00; oh[i01] = (_Float16)v01;
                     oh[i10] = (_Float16)v10; oh[i11] = (_Float16)v11; }
    if (flags & 8) {
      oh[(long)col0 * Mrows + row0] = (_Float16)v00;
      oh[(long)col1 * Mrows + row0] = (_Float16)v01;
      oh[(long)col0 * Mrows + row1] = (_Float16)v10;
      oh[(long)col1 * Mrows + row1] = (_Float16)v11;
    }
  }
}

// ---------------- attention: scores = Q * K^T (per b,h), K = HD = 32 (single step) -------
__global__ void k_gemm_qk(const _Float16* __restrict__ Q, const _Float16* __restrict__ Kx,
                          float* __restrict__ S) {
  const int lane = threadIdx.x;
  const int tj = blockIdx.x, ti = blockIdx.y, z = blockIdx.z;   // z = b*NH + h
  const int b = z >> 3, h = z & 7;
  const _Float16* Ap = Q + ((long)b * NT_ + ti * 16) * C_ + h * HD_;
  const _Float16* Bp = Kx + (long)b * NT_ * C_ + h * HD_ + (long)tj * 16 * C_;
  v8f acc = {0.f,0.f,0.f,0.f,0.f,0.f,0.f,0.f};
  const v16h af = frag_a_f16(Ap, C_, lane);
  const v16h bf = frag_b_k1(Bp, /*ldn(j)=*/C_, lane);   // d contiguous -> b128
  acc = wmma_f16(af, bf, acc);
  const int nl = lane & 15, g = (lane >> 4) & 1;
  float* out = S + (long)z * NT_ * NT_;
#pragma unroll
  for (int i = 0; i < 8; ++i)
    out[(long)(ti * 16 + g * 8 + i) * NT_ + tj * 16 + nl] = acc[i];
}

// ---------------- softmax over keys, scale + key-mask fused ----------------
__global__ void k_softmax(const float* __restrict__ S, const int* __restrict__ mask,
                          _Float16* __restrict__ P) {
  __shared__ float red[256];
  const int t = threadIdx.x;
  const long row = blockIdx.x;          // 0 .. B*NH*NT-1, layout [z][i]
  const int b = (int)(row >> 8) >> 3;
  float v = S[row * 256 + t] * SCALE_;
  if (mask && mask[b * 256 + t]) v = -__builtin_inff();
  red[t] = v; __syncthreads();
  for (int s = 128; s > 0; s >>= 1) { if (t < s) red[t] = fmaxf(red[t], red[t + s]); __syncthreads(); }
  const float mx = red[0]; __syncthreads();
  const float e = expf(v - mx);
  red[t] = e; __syncthreads();
  for (int s = 128; s > 0; s >>= 1) { if (t < s) red[t] += red[t + s]; __syncthreads(); }
  P[row * 256 + t] = (_Float16)(e / red[0]);
}

// ------- attention: out = P * V^T-layout (per b,h), 4x1 row-blocked, all contiguous -------
// Vt layout: vt[channel][token] = vt[(h*HD+d)*MR_ + b*NT_ + j]
__global__ void k_gemm_pv(const _Float16* __restrict__ P, const _Float16* __restrict__ Vt,
                          _Float16* __restrict__ O) {
  const int lane = threadIdx.x;
  const int td = blockIdx.x, tq = blockIdx.y, z = blockIdx.z;   // td: HD/16, tq: NT/64, z=b*NH+h
  const int b = z >> 3, h = z & 7;
  const _Float16* Ap = P + (long)z * NT_ * NT_ + (long)tq * 64 * NT_;
  const _Float16* Bp = Vt + (long)(h * HD_ + td * 16) * MR_ + b * NT_;  // B[k=j][n=d], j contig
  v8f acc[4];
#pragma unroll
  for (int s = 0; s < 4; ++s) acc[s] = (v8f){0.f,0.f,0.f,0.f,0.f,0.f,0.f,0.f};
  for (int k = 0; k < NT_; k += 32) {
    const v16h bf = frag_b_k1(Bp + k, /*ldn=*/MR_, lane);
#pragma unroll
    for (int s = 0; s < 4; ++s) {
      const v16h af = frag_a_f16(Ap + (long)s * 16 * NT_ + k, NT_, lane);
      acc[s] = wmma_f16(af, bf, acc[s]);
    }
  }
  const int nl = lane & 15, g = (lane >> 4) & 1;
#pragma unroll
  for (int s = 0; s < 4; ++s) {
#pragma unroll
    for (int i = 0; i < 8; ++i) {
      const long row = (long)tq * 64 + s * 16 + g * 8 + i;
      O[(long)b * NT_ * C_ + row * C_ + h * HD_ + td * 16 + nl] = (_Float16)acc[s][i];
    }
  }
}

// ------- 3x3 partial conv as implicit WMMA GEMM, 4x2 blocked (64 rows x 32 dc per wave).
// Weights pre-packed f16 as Wc[tap][dc][c] (c contiguous). Analytic border rescale + bias +
// exact GELU fused in the epilogue. -------
__global__ void k_conv3(const _Float16* __restrict__ X,   // (MR_, C_) pixel-normed, f16
                        const _Float16* __restrict__ Wc,  // [9][DC_][C_] f16
                        const float* __restrict__ B1,     // (DC_)
                        _Float16* __restrict__ Oh) {      // (MR_, DC_) f16 (GELU applied)
  const int lane = threadIdx.x;
  const int tn = blockIdx.x, tm = blockIdx.y;             // tn: DC/32, tm: MR/64
  const int r = lane & 15, g = (lane >> 4) & 1, nl = lane & 15;
  const int m0 = tm * 64;                                 // 64 rows stay inside one image
  const int b = m0 >> 8;
  v8f acc0[4], acc1[4];
#pragma unroll
  for (int s = 0; s < 4; ++s) {
    acc0[s] = (v8f){0.f,0.f,0.f,0.f,0.f,0.f,0.f,0.f};
    acc1[s] = acc0[s];
  }
  for (int tap = 0; tap < 9; ++tap) {
    const int dy = tap / 3 - 1, dx = tap % 3 - 1;
    bool valid[4]; const _Float16* src[4];
#pragma unroll
    for (int s = 0; s < 4; ++s) {
      const int n = (m0 + s * 16 + r) & 255;
      const int hh = (n >> 4) + dy, ww = (n & 15) + dx;
      valid[s] = ((unsigned)hh < 16u) && ((unsigned)ww < 16u);
      const int nsrc = valid[s] ? (hh * 16 + ww) : 0;
      src[s] = X + ((long)b * 256 + nsrc) * C_;
    }
    const _Float16* Wp0 = Wc + (long)tap * DC_ * C_ + (long)tn * 32 * C_;
    const _Float16* Wp1 = Wp0 + (long)16 * C_;
    for (int k = 0; k < C_; k += 32) {
      const v16h b0 = frag_b_k1(Wp0 + k, C_, lane);
      const v16h b1 = frag_b_k1(Wp1 + k, C_, lane);
#pragma unroll
      for (int s = 0; s < 4; ++s) {
        v16h af;
        const _Float16* p = src[s] + k + g * 8;
#pragma unroll
        for (int i = 0; i < 8; ++i) af[i]     = valid[s] ? p[i]      : (_Float16)0.f;
#pragma unroll
        for (int i = 0; i < 8; ++i) af[8 + i] = valid[s] ? p[16 + i] : (_Float16)0.f;
        acc0[s] = wmma_f16(af, b0, acc0[s]);
        acc1[s] = wmma_f16(af, b1, acc1[s]);
      }
    }
  }
  const int dc0 = tn * 32 + nl, dc1 = dc0 + 16;
  const float bv0 = B1[dc0], bv1 = B1[dc1];
#pragma unroll
  for (int s = 0; s < 4; ++s) {
#pragma unroll
    for (int i = 0; i < 8; ++i) {
      const int mr = m0 + s * 16 + g * 8 + i;
      const int nn = mr & 255, h2 = nn >> 4, w2 = nn & 15;
      const float sc = 9.0f / (float)((1 + (h2 > 0) + (h2 < 15)) * (1 + (w2 > 0) + (w2 < 15)));
      Oh[(long)mr * DC_ + dc0] = (_Float16)gelu_exact(acc0[s][i] * sc + bv0);
      Oh[(long)mr * DC_ + dc1] = (_Float16)gelu_exact(acc1[s][i] * sc + bv1);
    }
  }
}

// ---------------- pixel norm (+ optional pos-add / GELU), f32 in -> f16 out(s) ----------------
__global__ void k_pixelnorm(const float* __restrict__ X, const float* __restrict__ pos,
                            _Float16* __restrict__ o1, _Float16* __restrict__ o2, int do_gelu) {
  __shared__ float red[256];
  const int t = threadIdx.x;
  const long bn = blockIdx.x;
  const long idx = bn * 256 + t;
  const float v = X[idx];
  red[t] = v * v; __syncthreads();
  for (int s = 128; s > 0; s >>= 1) { if (t < s) red[t] += red[t + s]; __syncthreads(); }
  const float r = rsqrtf(red[0] * (1.0f / 256.0f) + 1e-8f);
  const float nv = v * r;
  if (o2) o2[idx] = (_Float16)(nv + pos[(long)(bn & 255) * 256 + t]);
  if (o1) o1[idx] = (_Float16)(do_gelu ? gelu_exact(nv) : nv);
}

// ---------------- weight staging kernels ----------------
// transpose+cast: Wt[n][k] = (f16)W[k][n]   (W is K rows x N cols, row-major)
__global__ void k_castT(const float* __restrict__ W, _Float16* __restrict__ Wt, int K, int N) {
  const long i = (long)blockIdx.x * 256 + threadIdx.x;   // over K*N outputs
  const int n = (int)(i / K), k = (int)(i % K);
  Wt[i] = (_Float16)W[(long)k * N + n];
}
// conv1 repack: Wc[tap][dc][c] = (f16)W1[dc][c][tap]
__global__ void k_cast_conv1(const float* __restrict__ W1, _Float16* __restrict__ Wc) {
  const long i = (long)blockIdx.x * 256 + threadIdx.x;   // over 9*DC*C outputs
  const int tap = (int)(i / ((long)DC_ * C_));
  const long r = i % ((long)DC_ * C_);
  const int dc = (int)(r / C_), c = (int)(r % C_);
  Wc[i] = (_Float16)W1[((long)dc * C_ + c) * 9 + tap];
}
// plain cast
__global__ void k_cast(const float* __restrict__ W, _Float16* __restrict__ Wh) {
  const long i = (long)blockIdx.x * 256 + threadIdx.x;
  Wh[i] = (_Float16)W[i];
}

// ---------------- misc elementwise kernels ----------------
__global__ void k_embed(const int* __restrict__ tok, const float* __restrict__ emb,
                        float* __restrict__ x0) {
  const long i = (long)blockIdx.x * 256 + threadIdx.x;   // B*N*C total
  const long bn = i >> 8; const int c = (int)(i & 255);
  x0[i] = emb[(long)tok[bn] * C_ + c];
}

__global__ void k_castpos(const float* __restrict__ M, const float* __restrict__ pos,
                          _Float16* __restrict__ mh, _Float16* __restrict__ mph) {
  const long i = (long)blockIdx.x * 256 + threadIdx.x;   // B*N*C total
  const float v = M[i];
  mh[i]  = (_Float16)v;
  mph[i] = (_Float16)(v + pos[i & (long)(NT_ * C_ - 1)]);
}

__global__ void k_masks(const float* __restrict__ mf, int* __restrict__ masks) {
  const int i = blockIdx.x * 256 + threadIdx.x;          // B*N total
  if (i >= B_ * NT_) return;
  float m = mf[i];
  for (int l = 0; l < 4; ++l) {
    masks[l * (B_ * NT_) + i]       = (m < 0.5f) ? 1 : 0;   // encoder key mask, layer l
    masks[(4 + l) * (B_ * NT_) + i] = (m < 0.2f) ? 1 : 0;   // decoder cross mask, exponent l
    m = sqrtf(m);
  }
}

__global__ void k_transpose_out(const float* __restrict__ Lg, float* __restrict__ out) {
  const long i = (long)blockIdx.x * 256 + threadIdx.x;   // B*V*N total, out is NCHW flat
  const int n = (int)(i & 255); const long bv = i >> 8;
  const int v = (int)(bv & (V_ - 1)); const int b = (int)(bv >> 10);
  out[i] = Lg[((long)b * 256 + n) * V_ + v];
}

// ---------------- host side ----------------
struct AttnW { const float *kb,*kw,*ob,*ow,*qb,*qw,*vb,*vw; };

struct Work {
  float *x0, *xcur, *xdec, *eout[4], *scores, *logits;
  _Float16 *s2h, *qkh, *qh, *kh, *vt, *attnh, *hnh, *hmid, *memh, *memph, *probsh;
  _Float16 *wq_t, *wk_t, *wv_t, *wo_t, *wc1, *wc2, *wout;   // f16 weight staging
  int *masks;
};

static void gemm_lin(hipStream_t s, const _Float16* A, int lda,
                     const _Float16* W, long ldn, const float* bias,
                     float* of, _Float16* oh, int M, int N, int K, int flags) {
  k_gemm_lin<<<dim3(N / 32, M / 32, 1), dim3(32), 0, s>>>(A, lda, W, ldn, bias, of, oh,
                                                          K, N, M, flags);
}

static void run_mha(hipStream_t s, Work& w, const _Float16* qin, const _Float16* kin,
                    const _Float16* vin, const AttnW& p, const int* mask, float* resid) {
  // stage weights: transpose + cast (C,C) -> [n][k] f16
  k_castT<<<dim3(C_ * C_ / 256), dim3(256), 0, s>>>(p.qw, w.wq_t, C_, C_);
  k_castT<<<dim3(C_ * C_ / 256), dim3(256), 0, s>>>(p.kw, w.wk_t, C_, C_);
  k_castT<<<dim3(C_ * C_ / 256), dim3(256), 0, s>>>(p.vw, w.wv_t, C_, C_);
  k_castT<<<dim3(C_ * C_ / 256), dim3(256), 0, s>>>(p.ow, w.wo_t, C_, C_);
  // projections (all-contiguous fragments); V written transposed for P*V
  gemm_lin(s, qin, C_, w.wq_t, C_, p.qb, nullptr, w.qh, MR_, C_, C_, 4);
  gemm_lin(s, kin, C_, w.wk_t, C_, p.kb, nullptr, w.kh, MR_, C_, C_, 4);
  gemm_lin(s, vin, C_, w.wv_t, C_, p.vb, nullptr, w.vt, MR_, C_, C_, 8);
  k_gemm_qk<<<dim3(NT_ / 16, NT_ / 16, B_ * NH_), dim3(32), 0, s>>>(w.qh, w.kh, w.scores);
  k_softmax<<<dim3(B_ * NH_ * NT_), dim3(256), 0, s>>>(w.scores, mask, w.probsh);
  k_gemm_pv<<<dim3(HD_ / 16, NT_ / 64, B_ * NH_), dim3(32), 0, s>>>(w.probsh, w.vt, w.attnh);
  // output projection, accumulated into residual
  gemm_lin(s, w.attnh, C_, w.wo_t, C_, p.ob, resid, nullptr, MR_, C_, C_, 1 | 2);
}

static void run_ffn(hipStream_t s, Work& w, float* x, const float* c1w, const float* c1b,
                    const float* c2w, const float* c2b) {
  k_pixelnorm<<<dim3(MR_), dim3(256), 0, s>>>(x, nullptr, w.hnh, nullptr, 0);
  k_cast_conv1<<<dim3(9 * DC_ * C_ / 256), dim3(256), 0, s>>>(c1w, w.wc1);
  k_conv3<<<dim3(DC_ / 32, MR_ / 64), dim3(32), 0, s>>>(w.hnh, w.wc1, c1b, w.hmid);
  // c2w shape (C_out=C, C_in=DC) is already [n][k]: straight cast, ldn = DC
  k_cast<<<dim3(C_ * DC_ / 256), dim3(256), 0, s>>>(c2w, w.wc2);
  gemm_lin(s, w.hmid, DC_, w.wc2, DC_, c2b, x, nullptr, MR_, C_, DC_, 1 | 2);
}

extern "C" void kernel_launch(void* const* d_in, const int* in_sizes, int n_in,
                              void* d_out, int out_size, void* d_ws, size_t ws_size,
                              hipStream_t stream) {
  (void)in_sizes; (void)n_in; (void)out_size; (void)ws_size;
  // -------- inputs: tokens, mask_f, then params leaves in JAX pytree (sorted-key) order ------
  const int*   tokens = (const int*)d_in[0];
  const float* mask_f = (const float*)d_in[1];
  int li = 2;
  auto F = [&]() { return (const float*)d_in[li++]; };
  auto readAttn = [&](AttnW& a) {
    a.kb = F(); a.kw = F(); a.ob = F(); a.ow = F();
    a.qb = F(); a.qw = F(); a.vb = F(); a.vw = F();
  };
  // params sorted keys: dec, embed, enc, out_b, out_w, pos
  AttnW dA[4], dX[4]; const float *dC1b[4], *dC1w[4], *dC2b[4], *dC2w[4];
  for (int l = 0; l < 4; ++l) {        // dec layer keys: attn, c1b, c1w, c2b, c2w, cross
    readAttn(dA[l]);
    dC1b[l] = F(); dC1w[l] = F(); dC2b[l] = F(); dC2w[l] = F();
    readAttn(dX[l]);
  }
  const float* embed = F();
  AttnW eA[4]; const float *eC1b[4], *eC1w[4], *eC2b[4], *eC2w[4];
  for (int l = 0; l < 4; ++l) {        // enc layer keys: attn, c1b, c1w, c2b, c2w
    readAttn(eA[l]);
    eC1b[l] = F(); eC1w[l] = F(); eC2b[l] = F(); eC2w[l] = F();
  }
  const float* out_b = F();
  const float* out_w = F();
  const float* pos   = F();

  // -------- workspace bump allocator (~38 MB) --------
  char* p = (char*)d_ws;
  auto alloc = [&](size_t bytes) -> void* {
    void* r = (void*)p; p += (bytes + 255) & ~(size_t)255; return r;
  };
  Work w;
  w.x0     = (float*)alloc((size_t)MR_ * C_ * 4);
  w.xcur   = (float*)alloc((size_t)MR_ * C_ * 4);
  w.xdec   = (float*)alloc((size_t)MR_ * C_ * 4);
  for (int l = 0; l < 4; ++l) w.eout[l] = (float*)alloc((size_t)MR_ * C_ * 4);
  w.scores = (float*)alloc((size_t)B_ * NH_ * NT_ * NT_ * 4);
  w.logits = (float*)alloc((size_t)MR_ * V_ * 4);
  w.s2h    = (_Float16*)alloc((size_t)MR_ * C_ * 2);
  w.qkh    = (_Float16*)alloc((size_t)MR_ * C_ * 2);
  w.qh     = (_Float16*)alloc((size_t)MR_ * C_ * 2);
  w.kh     = (_Float16*)alloc((size_t)MR_ * C_ * 2);
  w.vt     = (_Float16*)alloc((size_t)C_ * MR_ * 2);      // V transposed [channel][token]
  w.attnh  = (_Float16*)alloc((size_t)MR_ * C_ * 2);
  w.hnh    = (_Float16*)alloc((size_t)MR_ * C_ * 2);
  w.hmid   = (_Float16*)alloc((size_t)MR_ * DC_ * 2);
  w.memh   = (_Float16*)alloc((size_t)MR_ * C_ * 2);
  w.memph  = (_Float16*)alloc((size_t)MR_ * C_ * 2);
  w.probsh = (_Float16*)alloc((size_t)B_ * NH_ * NT_ * NT_ * 2);
  w.wq_t   = (_Float16*)alloc((size_t)C_ * C_ * 2);
  w.wk_t   = (_Float16*)alloc((size_t)C_ * C_ * 2);
  w.wv_t   = (_Float16*)alloc((size_t)C_ * C_ * 2);
  w.wo_t   = (_Float16*)alloc((size_t)C_ * C_ * 2);
  w.wc1    = (_Float16*)alloc((size_t)9 * DC_ * C_ * 2);
  w.wc2    = (_Float16*)alloc((size_t)C_ * DC_ * 2);
  w.wout   = (_Float16*)alloc((size_t)V_ * C_ * 2);
  w.masks  = (int*)alloc((size_t)8 * B_ * NT_ * 4);

  const size_t BNC = (size_t)MR_ * C_;

  // -------- embedding + masks --------
  k_embed<<<dim3((unsigned)(BNC / 256)), dim3(256), 0, stream>>>(tokens, embed, w.x0);
  k_masks<<<dim3(B_ * NT_ / 256), dim3(256), 0, stream>>>(mask_f, w.masks);

  // -------- encoder --------
  hipMemcpyAsync(w.xcur, w.x0, BNC * 4, hipMemcpyDeviceToDevice, stream);
  for (int l = 0; l < 4; ++l) {
    k_pixelnorm<<<dim3(MR_), dim3(256), 0, stream>>>(w.xcur, pos, w.s2h, w.qkh, 0);
    run_mha(stream, w, w.qkh, w.qkh, w.s2h, eA[l], w.masks + l * (B_ * NT_), w.xcur);
    run_ffn(stream, w, w.xcur, eC1w[l], eC1b[l], eC2w[l], eC2b[l]);
    hipMemcpyAsync(w.eout[l], w.xcur, BNC * 4, hipMemcpyDeviceToDevice, stream);
  }

  // -------- decoder --------
  hipMemcpyAsync(w.xdec, w.x0, BNC * 4, hipMemcpyDeviceToDevice, stream);
  for (int i = 0; i < 4; ++i) {
    const float* mem = w.eout[3 - i];
    const int* cmask = w.masks + (4 + (3 - i)) * (B_ * NT_);
    // self-attention (no mask)
    k_pixelnorm<<<dim3(MR_), dim3(256), 0, stream>>>(w.xdec, pos, w.s2h, w.qkh, 0);
    run_mha(stream, w, w.qkh, w.qkh, w.s2h, dA[i], nullptr, w.xdec);
    // cross-attention: q = t2+pos, k = mem+pos, v = mem
    k_pixelnorm<<<dim3(MR_), dim3(256), 0, stream>>>(w.xdec, pos, w.s2h, w.qkh, 0);
    k_castpos<<<dim3((unsigned)(BNC / 256)), dim3(256), 0, stream>>>(mem, pos, w.memh, w.memph);
    run_mha(stream, w, w.qkh, w.memph, w.memh, dX[i], cmask, w.xdec);
    run_ffn(stream, w, w.xdec, dC1w[i], dC1b[i], dC2w[i], dC2b[i]);
  }

  // -------- head: gelu(pixel_norm(out)) -> 1x1 conv to V logits, NCHW output --------
  k_pixelnorm<<<dim3(MR_), dim3(256), 0, stream>>>(w.xdec, nullptr, w.s2h, nullptr, 1);
  // out_w shape (V, C) is already [n][k]: straight cast, ldn = C
  k_cast<<<dim3(V_ * C_ / 256), dim3(256), 0, stream>>>(out_w, w.wout);
  gemm_lin(stream, w.s2h, C_, w.wout, C_, out_b, w.logits, nullptr, MR_, V_, C_, 1);
  k_transpose_out<<<dim3((unsigned)((size_t)B_ * V_ * NT_ / 256)), dim3(256), 0, stream>>>(
      w.logits, (float*)d_out);
}
